// CapsLayer_45681272160510
// MI455X (gfx1250) — compile-verified, matrix operand order checked by reading, not run on hardware
//
#include <hip/hip_runtime.h>
#include <hip/hip_bf16.h>
#include <math.h>

// Problem constants (match reference)
#define BATCH 32
#define NCAP  8192
#define PDIM  8
#define JCAP  16
#define DDIM  16
#define NC    64     // n per workgroup chunk (s-pass)
#define NC2   64     // n per workgroup chunk (agreement pass)
#define CSCALE 256.0f   // keeps c*w out of f16 subnormal range; undone at atomicAdd

typedef _Float16 v16h __attribute__((ext_vector_type(16)));
typedef _Float16 v8h  __attribute__((ext_vector_type(8)));
typedef float    v8f  __attribute__((ext_vector_type(8)));

__device__ __forceinline__ v16h combine16(v8h lo, v8h hi) {
    return __builtin_shufflevector(lo, hi, 0,1,2,3,4,5,6,7,8,9,10,11,12,13,14,15);
}

// ---------------------------------------------------------------------------
// init: b=0, c=1/N, s=0
// ---------------------------------------------------------------------------
__global__ void caps_init_kernel(float* __restrict__ bb, float* __restrict__ cc,
                                 float* __restrict__ s) {
    int i = blockIdx.x * blockDim.x + threadIdx.x;
    if (i < JCAP * NCAP) { bb[i] = 0.0f; cc[i] = 1.0f / (float)NCAP; }
    if (i < BATCH * JCAP * DDIM) s[i] = 0.0f;
}

__global__ void caps_zeros_kernel(float* __restrict__ s) {
    int i = blockIdx.x * blockDim.x + threadIdx.x;
    if (i < BATCH * JCAP * DDIM) s[i] = 0.0f;
}

// ---------------------------------------------------------------------------
// One-time f16 pre-convert into WMMA-operand-friendly layouts.
//   u16a[n][b][p]  (A-operand runs: 16B contiguous per (n,b))
//   w16t[j][n][d][p] (B-operand runs: 16B contiguous per (j,n,d); per-n 256B
//                     blocks contiguous -> async-copyable 1KB K-step tiles)
// ---------------------------------------------------------------------------
__global__ void caps_convu_kernel(const float* __restrict__ u,
                                  _Float16* __restrict__ u16a) {
    int idx = blockIdx.x * blockDim.x + threadIdx.x;   // idx = b*N + n
    if (idx >= BATCH * NCAP) return;
    int b = idx / NCAP, n = idx % NCAP;
    const float* src = u + (size_t)idx * PDIM;         // coalesced read
    _Float16* dst = u16a + ((size_t)n * BATCH + b) * PDIM;
    #pragma unroll
    for (int p = 0; p < PDIM; ++p) dst[p] = (_Float16)src[p];
}

__global__ void caps_convw_kernel(const float* __restrict__ w,
                                  _Float16* __restrict__ w16t) {
    int idx = blockIdx.x * blockDim.x + threadIdx.x;   // idx = j*N + n
    if (idx >= JCAP * NCAP) return;
    const float* src = w + (size_t)idx * (PDIM * DDIM);   // [p][d]
    _Float16* dst = w16t + (size_t)idx * (DDIM * PDIM);   // [d][p]
    for (int dg = 0; dg < 4; ++dg) {
        float4 r[8];
        #pragma unroll
        for (int p = 0; p < 8; ++p)
            r[p] = *(const float4*)&src[p * DDIM + dg * 4];
        #pragma unroll
        for (int dd = 0; dd < 4; ++dd) {
            v8h o;
            #pragma unroll
            for (int p = 0; p < 8; ++p) o[p] = (_Float16)((&r[p].x)[dd]);
            *(v8h*)&dst[(dg * 4 + dd) * PDIM] = o;
        }
    }
}

// ---------------------------------------------------------------------------
// s-pass (fast): s[b,j,d] += sum_n c[j,n]*u_hat[b,j,n,d], u_hat on the fly.
//   A operand: straight from global u16a (L2-resident, 16B runs, coalesced).
//   B operand: per-K-step 1KB w16t tile DMA'd to LDS with
//   GLOBAL_LOAD_ASYNC_TO_LDS_B128, double-buffered, s_wait_asynccnt pipelined.
//   c folded into B after the raw gather (2x v_pk_mul_f16 splats, x CSCALE).
//   No __syncthreads anywhere: waves fully independent.
// ---------------------------------------------------------------------------
__global__ __launch_bounds__(256) void caps_spass2_kernel(
    const _Float16* __restrict__ u16a,  // [N][B][P]
    const _Float16* __restrict__ w16t,  // [J][N][D][P]
    const float* __restrict__ cc,       // [J][N]
    float* __restrict__ s)              // [B][J][D]
{
    // per-wave double buffer: 2 x (4n * 16d * 8p) halves = 2 x 1024B
    __shared__ __align__(16) _Float16 btile[8 * 2 * 4 * DDIM * PDIM];

    const int tid  = threadIdx.x;
    const int wave = tid >> 5;
    const int lane = tid & 31;
    const int n0   = blockIdx.x * NC;
    const int m    = lane & 15;   // A-row / B-col / C-col
    const int hi   = lane >> 4;   // K-half selector (ISA 16-bit operand layout)

    _Float16* buf = &btile[wave * (2 * 4 * DDIM * PDIM)];
    const unsigned lds_base = (unsigned)(size_t)buf;   // low 32 bits = LDS offset

    for (int jj = 0; jj < 2; ++jj) {
        const int j = wave * 2 + jj;
        const _Float16* wsrc =
            w16t + ((size_t)j * NCAP + n0) * (DDIM * PDIM);
        v8f acc0 = {}, acc1 = {};

        // prologue: async-copy tile t=0 into buffer 0 (1KB, 2 x b128 per lane)
        {
            unsigned dst = lds_base + (unsigned)lane * 16u;
            unsigned long long srcA = (unsigned long long)(const void*)(wsrc + lane * 8);
            asm volatile("global_load_async_to_lds_b128 %0, %1, off"
                         :: "v"(dst), "v"(srcA) : "memory");
            asm volatile("global_load_async_to_lds_b128 %0, %1, off"
                         :: "v"(dst + 512u), "v"(srcA + 512ull) : "memory");
        }

        for (int t = 0; t < NC / 4; ++t) {
            if (t + 1 < NC / 4) {
                // issue next tile, then wait for the current one (2 in flight)
                unsigned dst = lds_base + (unsigned)((t + 1) & 1) * 1024u
                                        + (unsigned)lane * 16u;
                unsigned long long srcA = (unsigned long long)(const void*)
                    (wsrc + (t + 1) * (4 * DDIM * PDIM) + lane * 8);
                asm volatile("global_load_async_to_lds_b128 %0, %1, off"
                             :: "v"(dst), "v"(srcA) : "memory");
                asm volatile("global_load_async_to_lds_b128 %0, %1, off"
                             :: "v"(dst + 512u), "v"(srcA + 512ull) : "memory");
                asm volatile("s_wait_asynccnt 2" ::: "memory");
            } else {
                asm volatile("s_wait_asynccnt 0" ::: "memory");
            }

            const _Float16* bt = buf + (t & 1) * (4 * DDIM * PDIM);
            const int tn = t * 4;

            // fold softmax coefficient (x CSCALE) into B after raw-f16 gather
            float cv1 = cc[j * NCAP + n0 + tn + hi] * CSCALE;
            float cv2 = cc[j * NCAP + n0 + tn + 2 + hi] * CSCALE;
            _Float16 h1 = (_Float16)cv1, h2 = (_Float16)cv2;
            v8h s1 = {h1, h1, h1, h1, h1, h1, h1, h1};
            v8h s2 = {h2, h2, h2, h2, h2, h2, h2, h2};
            v8h b1 = *(const v8h*)&bt[((hi) * DDIM + m) * PDIM];
            v8h b2 = *(const v8h*)&bt[((2 + hi) * DDIM + m) * PDIM];
            v16h Bm = combine16(b1 * s1, b2 * s2);

            // A operands from global (K = 8*n_off + p; runs at n_off=hi, 2+hi)
            const _Float16* ub = u16a + (size_t)(n0 + tn) * (BATCH * PDIM);
            v16h A0 = combine16(*(const v8h*)&ub[((hi) * BATCH + m) * PDIM],
                                *(const v8h*)&ub[((2 + hi) * BATCH + m) * PDIM]);
            v16h A1 = combine16(*(const v8h*)&ub[((hi) * BATCH + m + 16) * PDIM],
                                *(const v8h*)&ub[((2 + hi) * BATCH + m + 16) * PDIM]);

            acc0 = __builtin_amdgcn_wmma_f32_16x16x32_f16(false, A0, false, Bm,
                                                          (short)0, acc0, false, false);
            acc1 = __builtin_amdgcn_wmma_f32_16x16x32_f16(false, A1, false, Bm,
                                                          (short)0, acc1, false, false);
        }

        // C layout: VGPR r, lanes0-15 -> M=r, lanes16-31 -> M=8+r; N = m.
        #pragma unroll
        for (int r = 0; r < 8; ++r) {
            int b0 = r + 8 * hi;
            atomicAdd(&s[((size_t)b0 * JCAP + j) * DDIM + m],
                      acc0[r] * (1.0f / CSCALE));
            atomicAdd(&s[((size_t)(b0 + 16) * JCAP + j) * DDIM + m],
                      acc1[r] * (1.0f / CSCALE));
        }
    }
}

// ---------------------------------------------------------------------------
// s-pass (fallback, f32 source, used only if ws_size can't hold f16 copies)
// ---------------------------------------------------------------------------
__global__ __launch_bounds__(256) void caps_spassf_kernel(
    const float* __restrict__ u,   // [B][N][P]
    const float* __restrict__ w,   // [J][N][P][D]
    const float* __restrict__ cc,  // [J][N]
    float* __restrict__ s)         // [B][J][D]
{
    __shared__ __align__(16) _Float16 u_lds[NC * BATCH * PDIM];     // [nl][b][p]
    __shared__ __align__(16) _Float16 cw_lds[8 * 4 * DDIM * PDIM];  // per-wave [n4][d][p]

    const int tid  = threadIdx.x;
    const int wave = tid >> 5;
    const int lane = tid & 31;
    const int n0   = blockIdx.x * NC;

    for (int it = 0; it < (NC * BATCH) / 256; ++it) {
        int pair = it * 256 + tid;          // pair = b*NC + nl
        int b  = pair / NC;
        int nl = pair % NC;
        const float* src = u + ((size_t)b * NCAP + (size_t)(n0 + nl)) * PDIM;
        _Float16* dst = &u_lds[(nl * BATCH + b) * PDIM];
        #pragma unroll
        for (int p = 0; p < PDIM; ++p) dst[p] = (_Float16)src[p];
    }
    __syncthreads();

    const int m  = lane & 15;
    const int hi = lane >> 4;
    _Float16* cw = &cw_lds[wave * (4 * DDIM * PDIM)];

    for (int jj = 0; jj < 2; ++jj) {
        const int j = wave * 2 + jj;
        v8f acc0 = {}, acc1 = {};
        for (int t = 0; t < NC / 4; ++t) {
            const int tn = t * 4;
            {
                int ni = lane >> 3;
                int p  = lane & 7;
                int n  = n0 + tn + ni;
                float cv = cc[j * NCAP + n] * CSCALE;
                const float* wsrc = w + (((size_t)j * NCAP + n) * PDIM + p) * DDIM;
                if (t + 1 < NC / 4)
                    __builtin_prefetch(wsrc + 4 * PDIM * DDIM, 0, 1);
                #pragma unroll
                for (int d = 0; d < DDIM; ++d)
                    cw[(ni * DDIM + d) * PDIM + p] = (_Float16)(cv * wsrc[d]);
            }
            asm volatile("s_wait_dscnt 0" ::: "memory");

            v16h A0 = combine16(*(const v8h*)&u_lds[((tn + hi) * BATCH + m) * PDIM],
                                *(const v8h*)&u_lds[((tn + 2 + hi) * BATCH + m) * PDIM]);
            v16h A1 = combine16(*(const v8h*)&u_lds[((tn + hi) * BATCH + m + 16) * PDIM],
                                *(const v8h*)&u_lds[((tn + 2 + hi) * BATCH + m + 16) * PDIM]);
            v16h Bm = combine16(*(const v8h*)&cw[((hi) * DDIM + m) * PDIM],
                                *(const v8h*)&cw[((2 + hi) * DDIM + m) * PDIM]);

            acc0 = __builtin_amdgcn_wmma_f32_16x16x32_f16(false, A0, false, Bm,
                                                          (short)0, acc0, false, false);
            acc1 = __builtin_amdgcn_wmma_f32_16x16x32_f16(false, A1, false, Bm,
                                                          (short)0, acc1, false, false);
        }
        #pragma unroll
        for (int r = 0; r < 8; ++r) {
            int b0 = r + 8 * hi;
            atomicAdd(&s[((size_t)b0 * JCAP + j) * DDIM + m], acc0[r] * (1.0f / CSCALE));
            atomicAdd(&s[((size_t)(b0 + 16) * JCAP + j) * DDIM + m], acc1[r] * (1.0f / CSCALE));
        }
    }
}

// ---------------------------------------------------------------------------
// squash: v = squash(s); writes f16 transposed v [J][D][B]; optional f32 out
// ---------------------------------------------------------------------------
__global__ void caps_squash_kernel(const float* __restrict__ s,
                                   _Float16* __restrict__ vt,
                                   float* __restrict__ out, int writeOut) {
    int idx = blockIdx.x * blockDim.x + threadIdx.x;   // idx = b*J + j
    if (idx >= BATCH * JCAP) return;
    int b = idx / JCAP, j = idx % JCAP;
    const float* sp = s + (size_t)idx * DDIM;
    float vloc[DDIM];
    float sq = 0.0f;
    #pragma unroll
    for (int d = 0; d < DDIM; ++d) { vloc[d] = sp[d]; sq += vloc[d] * vloc[d]; }
    float scale = (sq / (1.0f + sq)) * rsqrtf(sq + 1e-9f);
    #pragma unroll
    for (int d = 0; d < DDIM; ++d) {
        float v = vloc[d] * scale;
        vt[((size_t)j * DDIM + d) * BATCH + b] = (_Float16)v;
        if (writeOut) out[(size_t)idx * DDIM + d] = v;
    }
}

// ---------------------------------------------------------------------------
// agreement pass: b[j,n] += (1/B) * sum_{p,d} w[j,n,p,d] * G_n[p,d]
//   G_n = u[:,n,:]^T (PxB) @ v[:,j,:] (BxD): K = B = 32 -> one WMMA per 2 n's
// ---------------------------------------------------------------------------
__global__ __launch_bounds__(256) void caps_bpass_kernel(
    const float* __restrict__ u,       // [B][N][P]
    const float* __restrict__ w,       // [J][N][P][D] (fallback gather)
    const _Float16* __restrict__ w16t, // [J][N][D][P] (fast gather)
    const _Float16* __restrict__ vt,   // [J][D][B]
    float* __restrict__ bb,            // [J][N]
    int useW16)
{
    __shared__ __align__(16) _Float16 ut[NC2 * PDIM * BATCH];  // [nl][p][b]
    const int tid  = threadIdx.x;
    const int wave = tid >> 5;
    const int lane = tid & 31;
    const int n0   = blockIdx.x * NC2;

    {
        int nl = tid >> 3;   // 0..31
        int p  = tid & 7;
        for (int half = 0; half < NC2 / 32; ++half) {
            int nn = half * 32 + nl;
            for (int b = 0; b < BATCH; ++b)
                ut[(nn * PDIM + p) * BATCH + b] =
                    (_Float16)u[((size_t)b * NCAP + (size_t)(n0 + nn)) * PDIM + p];
        }
    }
    __syncthreads();

    const int m  = lane & 15;   // A-row (= 8*nsel + p) and B/C column d
    const int hi = lane >> 4;   // K-half selector (batch index split)

    for (int jj = 0; jj < 2; ++jj) {
        const int j = wave * 2 + jj;
        const _Float16* vb = vt + ((size_t)j * DDIM + m) * BATCH;
        v16h Bv = combine16(*(const v8h*)(vb + 8 * hi),
                            *(const v8h*)(vb + 16 + 8 * hi));

        for (int np = 0; np < NC2 / 2; ++np) {
            int nsel = m >> 3;
            int p    = m & 7;
            int nloc = np * 2 + nsel;
            const _Float16* ar = &ut[(nloc * PDIM + p) * BATCH];
            v16h Av = combine16(*(const v8h*)(ar + 8 * hi),
                                *(const v8h*)(ar + 16 + 8 * hi));
            v8f Gz = {};
            v8f G = __builtin_amdgcn_wmma_f32_16x16x32_f16(false, Av, false, Bv,
                                                           (short)0, Gz, false, false);
            // lane holds G[row = r + 8*hi][d = m] -> n = np*2 + hi, p = r
            int n = n0 + np * 2 + hi;
            float part = 0.0f;
            if (useW16) {
                v8h wv = *(const v8h*)&w16t[(((size_t)j * NCAP + n) * DDIM + m) * PDIM];
                #pragma unroll
                for (int r = 0; r < 8; ++r) part += (float)wv[r] * G[r];
            } else {
                const float* wp = w + ((size_t)j * NCAP + n) * PDIM * DDIM + m;
                #pragma unroll
                for (int r = 0; r < 8; ++r) part += wp[r * DDIM] * G[r];
            }
            part += __shfl_xor(part, 1);
            part += __shfl_xor(part, 2);
            part += __shfl_xor(part, 4);
            part += __shfl_xor(part, 8);
            if (m == 0)
                bb[(size_t)j * NCAP + n] += part * (1.0f / (float)BATCH);
        }
    }
}

// ---------------------------------------------------------------------------
// softmax over N (axis=1 of (J,N)) per j
// ---------------------------------------------------------------------------
__global__ __launch_bounds__(256) void caps_softmax_kernel(
    const float* __restrict__ bb, float* __restrict__ cc) {
    __shared__ float red[256];
    const int j = blockIdx.x, tid = threadIdx.x;
    const float* row = bb + (size_t)j * NCAP;
    float mx = -1e30f;
    for (int n = tid; n < NCAP; n += 256) mx = fmaxf(mx, row[n]);
    red[tid] = mx; __syncthreads();
    for (int st = 128; st > 0; st >>= 1) {
        if (tid < st) red[tid] = fmaxf(red[tid], red[tid + st]);
        __syncthreads();
    }
    float rmax = red[0]; __syncthreads();
    float sum = 0.0f;
    for (int n = tid; n < NCAP; n += 256) sum += __expf(row[n] - rmax);
    red[tid] = sum; __syncthreads();
    for (int st = 128; st > 0; st >>= 1) {
        if (tid < st) red[tid] += red[tid + st];
        __syncthreads();
    }
    float inv = 1.0f / red[0];
    for (int n = tid; n < NCAP; n += 256)
        cc[(size_t)j * NCAP + n] = __expf(row[n] - rmax) * inv;
}

// ---------------------------------------------------------------------------
extern "C" void kernel_launch(void* const* d_in, const int* in_sizes, int n_in,
                              void* d_out, int out_size, void* d_ws, size_t ws_size,
                              hipStream_t stream) {
    const float* u = (const float*)d_in[0];   // (B,1,N,P)
    const float* w = (const float*)d_in[1];   // (1,J,N,P,D)
    float* out = (float*)d_out;               // (B,J,1,D,1) = B*J*D floats

    // Workspace layout (floats first, then 16B-aligned f16 regions)
    float* s  = (float*)d_ws;                          // B*J*D
    float* bb = s + BATCH * JCAP * DDIM;               // J*N
    float* cc = bb + JCAP * NCAP;                      // J*N
    _Float16* vt = (_Float16*)(cc + JCAP * NCAP);      // J*D*B halves
    _Float16* u16a = vt + (size_t)JCAP * DDIM * BATCH;             // N*B*P halves
    _Float16* w16t = u16a + (size_t)NCAP * BATCH * PDIM;           // J*N*D*P halves

    const size_t need =
        (size_t)(BATCH * JCAP * DDIM + 2 * JCAP * NCAP) * sizeof(float) +
        ((size_t)JCAP * DDIM * BATCH + (size_t)NCAP * BATCH * PDIM +
         (size_t)JCAP * NCAP * DDIM * PDIM) * sizeof(_Float16);
    const int fast = (ws_size >= need) ? 1 : 0;

    if (fast) {
        caps_convu_kernel<<<(BATCH * NCAP + 255) / 256, 256, 0, stream>>>(u, u16a);
        caps_convw_kernel<<<(JCAP * NCAP + 255) / 256, 256, 0, stream>>>(w, w16t);
    }
    caps_init_kernel<<<(JCAP * NCAP + 255) / 256, 256, 0, stream>>>(bb, cc, s);

    // iter 0: c uniform -> s -> v0
    if (fast) caps_spass2_kernel<<<NCAP / NC, 256, 0, stream>>>(u16a, w16t, cc, s);
    else      caps_spassf_kernel<<<NCAP / NC, 256, 0, stream>>>(u, w, cc, s);
    caps_squash_kernel<<<2, 256, 0, stream>>>(s, vt, out, 0);

    for (int r = 1; r < 3; ++r) {
        caps_bpass_kernel<<<NCAP / NC2, 256, 0, stream>>>(u, w, w16t, vt, bb, fast);
        caps_softmax_kernel<<<JCAP, 256, 0, stream>>>(bb, cc);
        caps_zeros_kernel<<<(BATCH * JCAP * DDIM + 255) / 256, 256, 0, stream>>>(s);
        if (fast) caps_spass2_kernel<<<NCAP / NC, 256, 0, stream>>>(u16a, w16t, cc, s);
        else      caps_spassf_kernel<<<NCAP / NC, 256, 0, stream>>>(u, w, cc, s);
        caps_squash_kernel<<<2, 256, 0, stream>>>(s, vt, out, (r == 2) ? 1 : 0);
    }
}
